// Model_996432413564
// MI455X (gfx1250) — matrix operand order repaired and checked by reading, not compiled, not run on hardware
//
#include <hip/hip_runtime.h>
#include <math.h>

typedef __bf16 bf16_t;
typedef __attribute__((ext_vector_type(16))) __bf16 v16bf;
typedef __attribute__((ext_vector_type(8)))  __bf16 v8bf;
typedef __attribute__((ext_vector_type(8)))  float  v8f;

#define EPS 1e-8f

// ---------------- helpers ----------------
__device__ __forceinline__ float block_sum(float v, float* red) {
  const int tid = threadIdx.x, n = blockDim.x;
  red[tid] = v; __syncthreads();
  for (int s = n >> 1; s > 0; s >>= 1) {
    if (tid < s) red[tid] += red[tid + s];
    __syncthreads();
  }
  float r = red[0]; __syncthreads();
  return r;
}

__global__ __launch_bounds__(256) void fill_zero(float* __restrict__ p, unsigned long long n) {
  unsigned long long i = (unsigned long long)blockIdx.x * blockDim.x + threadIdx.x;
  if (i < n) p[i] = 0.f;
}

__global__ __launch_bounds__(256) void cvt_bf16(const float* __restrict__ in,
                                                bf16_t* __restrict__ out, int n) {
  int i = blockIdx.x * blockDim.x + threadIdx.x;
  if (i < n) out[i] = (bf16_t)in[i];
}

// W (K x N, row major, f32) -> Wt (N x K, row major, bf16)
__global__ __launch_bounds__(256) void transpose_cvt(const float* __restrict__ W,
                                                     bf16_t* __restrict__ Wt,
                                                     int K, int N) {
  __shared__ float t[32][33];
  const int tx = threadIdx.x & 31, ty = threadIdx.x >> 5;  // 256 threads
  const int n0 = blockIdx.x * 32, k0 = blockIdx.y * 32;
  for (int i = 0; i < 32; i += 8)
    t[ty + i][tx] = W[(size_t)(k0 + ty + i) * N + (n0 + tx)];
  __syncthreads();
  for (int i = 0; i < 32; i += 8)
    Wt[(size_t)(n0 + ty + i) * K + (k0 + tx)] = (bf16_t)t[tx][ty + i];
}

// ---------------- WMMA GEMM ----------------
// Y = A(M x K, bf16 row-major) @ Wt^T (Wt is N x K bf16 row-major) + bias
// Block: 256 threads = 8 waves; tile = MR rows x 512 cols (wave owns 64 cols).
// K is a template constant: every fragment load is (base pointer + compile-time
// immediate offset), the K-loop fully unrolls, and fragments live in a 2-deep
// compile-time-indexed buffer (no pointer arrays, no register rotation movs).
// EPI 0: fp32 out + bias.  EPI 1: bf16 out + bias.
// EPI 2: bias + LayerNorm(row, 512) + leaky-relu -> bf16 out (MR==16, Ntot==512).
template <int EPI, int MR, int K>
__global__ __launch_bounds__(256) void gemm_wmma(
    const bf16_t* __restrict__ A, const bf16_t* __restrict__ Wt,
    const float* __restrict__ bias,
    const float* __restrict__ g, const float* __restrict__ bt,
    float* __restrict__ outF, bf16_t* __restrict__ outH,
    int Ntot, float slope) {
  static_assert(EPI != 2 || MR == 16, "LN epilogue requires MR==16");
  constexpr int AT = MR / 16;
  constexpr int NSTEP = K / 32;
  const int tid  = threadIdx.x;
  const int wave = tid >> 5, lane = tid & 31;
  const int r16 = lane & 15, hh = lane >> 4;
  const int m0 = blockIdx.x * MR;
  const int nwbase = blockIdx.y * 512 + wave * 64;

  v8f acc[AT][4];
#pragma unroll
  for (int at = 0; at < AT; at++)
#pragma unroll
    for (int nt = 0; nt < 4; nt++)
#pragma unroll
      for (int i = 0; i < 8; i++) acc[at][nt][i] = 0.f;

  // single base pointers; all fragment addresses are compile-time immediates off these
  const bf16_t* aBase = A  + (size_t)(m0 + r16) * K + hh * 8;
  const bf16_t* bBase = Wt + (size_t)(nwbase + r16) * K + hh * 8;

  // A fragment (ISA 16-bit A 16x32): lanes 0-15 K=0..7,16..23 ; lanes 16-31 K=8..15,24..31
  auto loadA = [&](v16bf& f, int at, int s) {
    v8bf lo = *(const v8bf*)(aBase + at * 16 * K + s * 32);
    v8bf hi = *(const v8bf*)(aBase + at * 16 * K + s * 32 + 16);
#pragma unroll
    for (int i = 0; i < 8; i++) { f[i] = lo[i]; f[8 + i] = hi[i]; }
  };
  // B fragment (16-bit B 32x16): lanes 0-15 K=0..7,16..23 ; lanes 16-31 K=8..15,24..31
  auto loadB = [&](v16bf& f, int nt, int s) {
    v8bf lo = *(const v8bf*)(bBase + nt * 16 * K + s * 32);
    v8bf hi = *(const v8bf*)(bBase + nt * 16 * K + s * 32 + 16);
#pragma unroll
    for (int i = 0; i < 8; i++) { f[i] = lo[i]; f[8 + i] = hi[i]; }
  };

  v16bf aF[2][AT], bF[2][4];
#pragma unroll
  for (int at = 0; at < AT; at++) loadA(aF[0][at], at, 0);
#pragma unroll
  for (int nt = 0; nt < 4; nt++) loadB(bF[0][nt], nt, 0);

#pragma unroll
  for (int s = 0; s < NSTEP; s++) {
    const int cur = s & 1, nxt = cur ^ 1;
    if (s + 1 < NSTEP) {
#pragma unroll
      for (int at = 0; at < AT; at++) loadA(aF[nxt][at], at, s + 1);
#pragma unroll
      for (int nt = 0; nt < 4; nt++) loadB(bF[nxt][nt], nt, s + 1);
    }
#pragma unroll
    for (int nt = 0; nt < 4; nt++)
#pragma unroll
      for (int at = 0; at < AT; at++)
        acc[at][nt] = __builtin_amdgcn_wmma_f32_16x16x32_bf16(
            false, aF[cur][at], false, bF[cur][nt], (short)0, acc[at][nt], false, false);
  }

  if constexpr (EPI == 0 || EPI == 1) {
#pragma unroll
    for (int at = 0; at < AT; at++) {
#pragma unroll
      for (int nt = 0; nt < 4; nt++) {
        const int col = nwbase + nt * 16 + r16;
        const float bv = bias ? bias[col] : 0.f;
#pragma unroll
        for (int v = 0; v < 8; v++) {
          const int row = m0 + at * 16 + v + 8 * hh;  // C/D: lanes 16-31 hold rows +8
          const float y = acc[at][nt][v] + bv;
          if constexpr (EPI == 0) outF[(size_t)row * Ntot + col] = y;
          else                    outH[(size_t)row * Ntot + col] = (bf16_t)y;
        }
      }
    }
  } else {
    __shared__ float tile[16][513];
    __shared__ float ps[16][16], ps2[16][16];
    __shared__ float mus[16], rstds[16];
#pragma unroll
    for (int nt = 0; nt < 4; nt++) {
      const int col = nwbase + nt * 16 + r16;  // blockIdx.y == 0 here
      const float bv = bias[col];
#pragma unroll
      for (int v = 0; v < 8; v++) tile[v + 8 * hh][col] = acc[0][nt][v] + bv;
    }
    __syncthreads();
    const int row = tid >> 4, seg = tid & 15;
    float s = 0.f, s2 = 0.f;
#pragma unroll
    for (int c0 = 0; c0 < 32; c0++) {
      float x = tile[row][seg * 32 + c0];
      s += x; s2 += x * x;
    }
    ps[row][seg] = s; ps2[row][seg] = s2;
    __syncthreads();
    if (tid < 16) {
      float ss = 0.f, ss2 = 0.f;
#pragma unroll
      for (int i = 0; i < 16; i++) { ss += ps[tid][i]; ss2 += ps2[tid][i]; }
      float mu = ss * (1.f / 512.f);
      float var = ss2 * (1.f / 512.f) - mu * mu;
      mus[tid] = mu;
      rstds[tid] = rsqrtf(var + 1e-5f);
    }
    __syncthreads();
    const float mu = mus[row], rs = rstds[row];
#pragma unroll
    for (int c0 = 0; c0 < 32; c0++) {
      int c = seg * 32 + c0;
      float x = (tile[row][c] - mu) * rs * g[c] + bt[c];
      x = x > 0.f ? x : slope * x;
      outH[(size_t)(m0 + row) * 512 + c] = (bf16_t)x;
    }
  }
}

// ---------------- stage kernels ----------------
// D (256 x 65536 bf16, layout [r][c*128+t]) -> unit-norm over c, transposed
// Dn[(r*128+t)*512+c] bf16 ; leak lk[r*128+t] = 0.1 + 0.98*sigmoid(dn . W_l + b_l)
__global__ __launch_bounds__(128) void deform_norm(
    const bf16_t* __restrict__ D, bf16_t* __restrict__ Dn,
    float* __restrict__ lk, const float* __restrict__ Wl,
    const float* __restrict__ bl) {
  __shared__ float red[128];
  const int rt = blockIdx.x;          // r*128 + t
  const int r = rt >> 7, t = rt & 127;
  const int tid = threadIdx.x;        // 128 threads
  const bf16_t* base = D + (size_t)r * 65536 + t;
  float v[4]; float s2 = 0.f;
  for (int j = 0; j < 4; j++) {
    int c = tid + j * 128;
    float x = (float)base[(size_t)c * 128];
    v[j] = x; s2 += x * x;
  }
  float nrm = sqrtf(block_sum(s2, red)) + EPS;
  float dot = 0.f;
  for (int j = 0; j < 4; j++) {
    int c = tid + j * 128;
    float dn = v[j] / nrm;
    Dn[(size_t)rt * 512 + c] = (bf16_t)dn;
    dot += dn * Wl[c];
  }
  float dsum = block_sum(dot, red);
  if (tid == 0) lk[rt] = 0.1f + 0.98f / (1.f + expf(-(dsum + bl[0])));
}

// impulse (r x 4096) -> windows (r*16+f rows of 512, hop 256, zero-padded) + row norms
__global__ __launch_bounds__(128) void impulse_window(
    const float* __restrict__ imp, bf16_t* __restrict__ win, float* __restrict__ nrm) {
  __shared__ float red[128];
  const int row = blockIdx.x;  // r*16+f
  const int r = row >> 4, f = row & 15;
  const int tid = threadIdx.x;
  float s2 = 0.f;
  for (int j = 0; j < 4; j++) {
    int c = tid + j * 128;
    int idx = f * 256 + c;
    float x = (idx < 4096) ? imp[(size_t)r * 4096 + idx] : 0.f;
    s2 += x * x;
    win[(size_t)row * 512 + c] = (bf16_t)x;
  }
  float t = block_sum(s2, red);
  if (tid == 0) nrm[row] = sqrtf(t);
}

// out_row = X_row / (||X_row||+EPS) * scale[row]   (fp32 out)
__global__ __launch_bounds__(128) void rownorm_scale(
    const float* __restrict__ X, const float* __restrict__ sc, float* __restrict__ out) {
  __shared__ float red[128];
  const int row = blockIdx.x;
  const int tid = threadIdx.x;
  float v[4]; float s2 = 0.f;
  for (int j = 0; j < 4; j++) {
    int c = tid + j * 128;
    float x = X[(size_t)row * 512 + c];
    v[j] = x; s2 += x * x;
  }
  float nrm = sqrtf(block_sum(s2, red)) + EPS;
  float scale = sc[row] / nrm;
  for (int j = 0; j < 4; j++) {
    int c = tid + j * 128;
    out[(size_t)row * 512 + c] = v[j] * scale;
  }
}

// scan step, phase 1: h += e ; cn = ||h|| ; h = h*w + b ; h = unit(h)*cn
__global__ __launch_bounds__(256) void scan_pre(
    float* __restrict__ h, const float* __restrict__ iemb,
    const bf16_t* __restrict__ wgt, const bf16_t* __restrict__ bia,
    bf16_t* __restrict__ hx, float* __restrict__ cn, int t) {
  __shared__ float red[256];
  const int r = blockIdx.x;
  const int tid = threadIdx.x;  // 256 threads x 2 elems
  float v[2]; float s2 = 0.f;
  for (int j = 0; j < 2; j++) {
    int c = tid + j * 256;
    float x = h[(size_t)r * 512 + c];
    if (t < 16) x += iemb[((size_t)(r * 16 + t)) * 512 + c];
    v[j] = x; s2 += x * x;
  }
  float cnv = sqrtf(block_sum(s2, red));
  const size_t wb = ((size_t)r * 128 + t) * 512;
  float s2b = 0.f;
  for (int j = 0; j < 2; j++) {
    int c = tid + j * 256;
    // wgt/bia slices are streamed exactly once -> non-temporal loads
    float w = (float)__builtin_nontemporal_load(wgt + wb + c);
    float b = (float)__builtin_nontemporal_load(bia + wb + c);
    float y = v[j] * w + b;
    v[j] = y; s2b += y * y;
  }
  float nn = sqrtf(block_sum(s2b, red)) + EPS;
  for (int j = 0; j < 2; j++) {
    int c = tid + j * 256;
    float hn = v[j] / nn * cnv;
    h[(size_t)r * 512 + c] = hn;
    hx[(size_t)r * 512 + c] = (bf16_t)hn;
  }
  if (tid == 0) cn[r] = cnv;
}

// phase 2: ol = unit(ol_raw)*(cn*lk) ; on = ||ol|| ; h = unit(h-ol)*(cn-on)
__global__ __launch_bounds__(256) void scan_mid(
    float* __restrict__ h, const float* __restrict__ olraw,
    const float* __restrict__ cn, const float* __restrict__ lk,
    bf16_t* __restrict__ olx, float* __restrict__ onv, int t) {
  __shared__ float red[256];
  const int r = blockIdx.x;
  const int tid = threadIdx.x;
  float o[2]; float s2 = 0.f;
  for (int j = 0; j < 2; j++) {
    int c = tid + j * 256;
    float x = olraw[(size_t)r * 512 + c];
    o[j] = x; s2 += x * x;
  }
  float no = sqrtf(block_sum(s2, red));
  float cnv = cn[r];
  float scale = cnv * lk[(size_t)r * 128 + t] / (no + EPS);
  float onval = scale * no;
  float v[2]; float s2b = 0.f;
  for (int j = 0; j < 2; j++) {
    int c = tid + j * 256;
    float ol = o[j] * scale;
    olx[(size_t)r * 512 + c] = (bf16_t)ol;
    float hn = h[(size_t)r * 512 + c] - ol;
    v[j] = hn; s2b += hn * hn;
  }
  float nh = sqrtf(block_sum(s2b, red)) + EPS;
  float tgt = cnv - onval;
  for (int j = 0; j < 2; j++) {
    int c = tid + j * 256;
    h[(size_t)r * 512 + c] = v[j] / nh * tgt;
  }
  if (tid == 0) onv[r] = onval;
}

// phase 3: of = unit(of_raw * hamming)*on ; overlap-add into output at hop 256
__global__ __launch_bounds__(256) void scan_post(
    const float* __restrict__ ofraw, const float* __restrict__ onv,
    float* __restrict__ out, int t) {
  __shared__ float red[256];
  const int r = blockIdx.x;
  const int tid = threadIdx.x;
  float v[2]; float s2 = 0.f;
  for (int j = 0; j < 2; j++) {
    int c = tid + j * 256;
    float ham = 0.54f - 0.46f * cosf(6.28318530718f * (float)c / 512.f);
    float x = ofraw[(size_t)r * 512 + c] * ham;
    v[j] = x; s2 += x * x;
  }
  float nrm = sqrtf(block_sum(s2, red)) + EPS;
  float sc = onv[r] / nrm;
  for (int j = 0; j < 2; j++) {
    int c = tid + j * 256;
    int pos = t * 256 + c;
    if (pos < 32768) out[(size_t)r * 32768 + pos] += v[j] * sc;
  }
}

// ---------------- host ----------------
extern "C" void kernel_launch(void* const* d_in, const int* in_sizes, int n_in,
                              void* d_out, int out_size, void* d_ws, size_t ws_size,
                              hipStream_t stream) {
  const float* embedding = (const float*)d_in[0];   // (4,64,256)
  const float* impulse   = (const float*)d_in[1];   // (4,64,4096)
  const float* W_def = (const float*)d_in[2];       // (256,65536)
  const float* b_def = (const float*)d_in[3];       // (65536)
  const float* W_w = (const float*)d_in[4];         // (512,512)
  const float* b_w = (const float*)d_in[5];
  const float* W_b = (const float*)d_in[6];
  const float* b_b = (const float*)d_in[7];
  const float* W_l = (const float*)d_in[8];         // (512,1)
  const float* b_l = (const float*)d_in[9];         // (1)
  const float* Ws_imp = (const float*)d_in[10];     // (4,512,512)
  const float* bs_imp = (const float*)d_in[11];     // (4,512)
  const float* g_imp  = (const float*)d_in[12];     // (3,512)
  const float* bt_imp = (const float*)d_in[13];
  const float* Ws_lat = (const float*)d_in[14];
  const float* bs_lat = (const float*)d_in[15];
  const float* g_lat  = (const float*)d_in[16];
  const float* bt_lat = (const float*)d_in[17];
  const float* Ws_out = (const float*)d_in[18];
  const float* bs_out = (const float*)d_in[19];
  const float* g_out  = (const float*)d_in[20];
  const float* bt_out = (const float*)d_in[21];
  (void)in_sizes; (void)n_in; (void)out_size;

  char* ws = (char*)d_ws;
  size_t off = 0;
  auto carve = [&](size_t bytes) -> char* {
    char* p = ws + off;
    off += (bytes + 255) & ~(size_t)255;
    return p;
  };

  bf16_t* Wdef_t = (bf16_t*)carve(65536ull * 256 * 2);   // 32 MB
  bf16_t* Dh     = (bf16_t*)carve(256ull * 65536 * 2);   // 32 MB
  bf16_t* Dn     = (bf16_t*)carve(32768ull * 512 * 2);   // 32 MB
  bf16_t* wgt    = (bf16_t*)carve(32768ull * 512 * 2);   // 32 MB
  bf16_t* bia    = (bf16_t*)carve(32768ull * 512 * 2);   // 32 MB
  bf16_t* Wt_w   = (bf16_t*)carve(512ull * 512 * 2);
  bf16_t* Wt_b   = (bf16_t*)carve(512ull * 512 * 2);
  bf16_t* Wt_imp = (bf16_t*)carve(4ull * 512 * 512 * 2);
  bf16_t* Wt_lat = (bf16_t*)carve(4ull * 512 * 512 * 2);
  bf16_t* Wt_out = (bf16_t*)carve(4ull * 512 * 512 * 2);
  bf16_t* emb_h  = (bf16_t*)carve(256ull * 256 * 2);
  float*  lk     = (float*)carve(32768ull * 4);
  bf16_t* win    = (bf16_t*)carve(4096ull * 512 * 2);
  float*  wnorm  = (float*)carve(4096ull * 4);
  bf16_t* actA   = (bf16_t*)carve(4096ull * 512 * 2);
  bf16_t* actB   = (bf16_t*)carve(4096ull * 512 * 2);
  float*  raw    = (float*)carve(4096ull * 512 * 4);
  float*  iemb   = (float*)carve(4096ull * 512 * 4);
  float*  hst    = (float*)carve(256ull * 512 * 4);
  bf16_t* hx     = (bf16_t*)carve(256ull * 512 * 2);
  bf16_t* olx    = (bf16_t*)carve(256ull * 512 * 2);
  float*  cn     = (float*)carve(256ull * 4);
  float*  onv    = (float*)carve(256ull * 4);
  if (off > ws_size) return;  // workspace too small (needs ~206 MB)

  float* out = (float*)d_out;  // (4,64,32768) fp32

  // zero output (overlap-add target) and h state
  fill_zero<<<32768, 256, 0, stream>>>(out, 8388608ull);
  fill_zero<<<512, 256, 0, stream>>>(hst, 131072ull);

  // weight conversion / transposition (f32 -> bf16 N x K)
  cvt_bf16<<<256, 256, 0, stream>>>(embedding, emb_h, 65536);
  transpose_cvt<<<dim3(2048, 8), 256, 0, stream>>>(W_def, Wdef_t, 256, 65536);
  transpose_cvt<<<dim3(16, 16), 256, 0, stream>>>(W_w, Wt_w, 512, 512);
  transpose_cvt<<<dim3(16, 16), 256, 0, stream>>>(W_b, Wt_b, 512, 512);
  for (int j = 0; j < 4; j++) {
    size_t o = (size_t)j * 512 * 512;
    transpose_cvt<<<dim3(16, 16), 256, 0, stream>>>(Ws_imp + o, Wt_imp + o, 512, 512);
    transpose_cvt<<<dim3(16, 16), 256, 0, stream>>>(Ws_lat + o, Wt_lat + o, 512, 512);
    transpose_cvt<<<dim3(16, 16), 256, 0, stream>>>(Ws_out + o, Wt_out + o, 512, 512);
  }

  // Stage 1: deform = emb @ W_def + b_def  (M=256, N=65536, K=256), 32-row tiles
  gemm_wmma<1, 32, 256><<<dim3(8, 128), 256, 0, stream>>>(
      emb_h, Wdef_t, b_def, nullptr, nullptr, nullptr, Dh, 65536, 0.f);
  // unit-norm over channels + transpose to (r,t,c) + leak
  deform_norm<<<32768, 128, 0, stream>>>(Dh, Dn, lk, W_l, b_l);

  // Stage 2: weights / biases GEMMs (M=32768, N=512, K=512), 32-row tiles
  gemm_wmma<1, 32, 512><<<dim3(1024, 1), 256, 0, stream>>>(
      Dn, Wt_w, b_w, nullptr, nullptr, nullptr, wgt, 512, 0.f);
  gemm_wmma<1, 32, 512><<<dim3(1024, 1), 256, 0, stream>>>(
      Dn, Wt_b, b_b, nullptr, nullptr, nullptr, bia, 512, 0.f);

  // Stage 3: impulse windows + 4-layer MLP + unit-norm * window-norm
  impulse_window<<<4096, 128, 0, stream>>>(impulse, win, wnorm);
  gemm_wmma<2, 16, 512><<<dim3(256, 1), 256, 0, stream>>>(
      win, Wt_imp + 0 * 262144, bs_imp + 0, g_imp + 0, bt_imp + 0, nullptr, actA, 512, 0.2f);
  gemm_wmma<2, 16, 512><<<dim3(256, 1), 256, 0, stream>>>(
      actA, Wt_imp + 1 * 262144, bs_imp + 512, g_imp + 512, bt_imp + 512, nullptr, actB, 512, 0.2f);
  gemm_wmma<2, 16, 512><<<dim3(256, 1), 256, 0, stream>>>(
      actB, Wt_imp + 2 * 262144, bs_imp + 1024, g_imp + 1024, bt_imp + 1024, nullptr, actA, 512, 0.2f);
  gemm_wmma<0, 32, 512><<<dim3(128, 1), 256, 0, stream>>>(
      actA, Wt_imp + 3 * 262144, bs_imp + 1536, nullptr, nullptr, raw, nullptr, 512, 0.f);
  rownorm_scale<<<4096, 128, 0, stream>>>(raw, wnorm, iemb);

  // Stage 4: sequential 128-step scan (stream order gives step ordering).
  // MR=16 here: M=256 only -> 16 blocks keeps more WGPs busy (latency-bound phase).
  for (int t = 0; t < 128; t++) {
    scan_pre<<<256, 256, 0, stream>>>(hst, iemb, wgt, bia, hx, cn, t);
    gemm_wmma<2, 16, 512><<<dim3(16, 1), 256, 0, stream>>>(
        hx, Wt_lat + 0 * 262144, bs_lat + 0, g_lat + 0, bt_lat + 0, nullptr, actA, 512, 0.2f);
    gemm_wmma<2, 16, 512><<<dim3(16, 1), 256, 0, stream>>>(
        actA, Wt_lat + 1 * 262144, bs_lat + 512, g_lat + 512, bt_lat + 512, nullptr, actB, 512, 0.2f);
    gemm_wmma<2, 16, 512><<<dim3(16, 1), 256, 0, stream>>>(
        actB, Wt_lat + 2 * 262144, bs_lat + 1024, g_lat + 1024, bt_lat + 1024, nullptr, actA, 512, 0.2f);
    gemm_wmma<0, 16, 512><<<dim3(16, 1), 256, 0, stream>>>(
        actA, Wt_lat + 3 * 262144, bs_lat + 1536, nullptr, nullptr, raw, nullptr, 512, 0.f);
    scan_mid<<<256, 256, 0, stream>>>(hst, raw, cn, lk, olx, onv, t);
    gemm_wmma<2, 16, 512><<<dim3(16, 1), 256, 0, stream>>>(
        olx, Wt_out + 0 * 262144, bs_out + 0, g_out + 0, bt_out + 0, nullptr, actA, 512, 0.2f);
    gemm_wmma<2, 16, 512><<<dim3(16, 1), 256, 0, stream>>>(
        actA, Wt_out + 1 * 262144, bs_out + 512, g_out + 512, bt_out + 512, nullptr, actB, 512, 0.2f);
    gemm_wmma<2, 16, 512><<<dim3(16, 1), 256, 0, stream>>>(
        actB, Wt_out + 2 * 262144, bs_out + 1024, g_out + 1024, bt_out + 1024, nullptr, actA, 512, 0.2f);
    gemm_wmma<0, 16, 512><<<dim3(16, 1), 256, 0, stream>>>(
        actA, Wt_out + 3 * 262144, bs_out + 1536, nullptr, nullptr, raw, nullptr, 512, 0.f);
    scan_post<<<256, 256, 0, stream>>>(raw, onv, out, t);
  }
}
